// PoiExtractor_40853728920096
// MI455X (gfx1250) — matrix operand order repaired and checked by reading, not compiled, not run on hardware
//
#include <hip/hip_runtime.h>

// ---------------------------------------------------------------------------
// Types for CDNA5 WMMA (wave32, 16x16x32 bf16 -> f32)
// ---------------------------------------------------------------------------
typedef __attribute__((ext_vector_type(16))) __bf16 v16bf;
typedef __attribute__((ext_vector_type(8)))  float  v8f;
typedef __attribute__((ext_vector_type(4)))  float  f32x4;

union FragBf { f32x4 f[2]; v16bf v; };

// Problem sizes (fixed by the reference)
#define BB 8
#define CC 128
#define HH 256
#define WW 256
#define NB 200
#define SS 256
#define MTOT (BB * NB * SS)          // 409600 points
#define HETER_ELEMS (BB * CC * HH * WW)  // 67108864
#define PRED_ELEMS  (MTOT * CC)          // 52428800

// LDS activation buffer leading dim (bf16 elems); 264*2B = 528B row pitch
// (not a multiple of the 256B LDS bank period -> conflict-friendly)
#define LDA 264

// Weight (bf16, transposed [N][Kpad]) offsets inside d_ws weight region
#define OFF_W0 0              // 256 x 64
#define OFF_W1 16384          // 256 x 256
#define OFF_W2 81920
#define OFF_W3 147456
#define OFF_W4 212992
#define OFF_W5 278528         // 128 x 256
#define WT_TOTAL 311296

__device__ __forceinline__ unsigned short f2bf(float x) {
    union { float f; unsigned int u; } c; c.f = x;
    unsigned int r = c.u + 0x7FFFu + ((c.u >> 16) & 1u);  // round-to-nearest-even
    return (unsigned short)(r >> 16);
}

__device__ __forceinline__ v8f zero_v8f() {
    v8f z = {0.f, 0.f, 0.f, 0.f, 0.f, 0.f, 0.f, 0.f};
    return z;
}

// ---------------------------------------------------------------------------
// Weight prep: f32 [K][N] (row-major) -> bf16 W^T [N][Kpad] (K contiguous),
// zero-padded in K so B-fragments load as two contiguous 16B chunks per lane.
// ---------------------------------------------------------------------------
__global__ void convert_weight_kernel(const float* __restrict__ src,
                                      unsigned short* __restrict__ dst,
                                      int K, int N, int Kpad) {
    int idx = blockIdx.x * 256 + threadIdx.x;
    if (idx >= N * Kpad) return;
    int n = idx / Kpad;
    int k = idx - n * Kpad;
    float v = (k < K) ? src[(size_t)k * N + n] : 0.0f;
    dst[idx] = f2bf(v);
}

// ---------------------------------------------------------------------------
// Gather kernel: one block per box (b,n); thread = sample s.
// Computes the box transform, validity mask (stored to d_ws for the MLP),
// and the 128-channel bilinear sample -> gt. 256 samples of a box cover a
// ~6x3 cell neighborhood, so taps hit hot L2 lines.
// ---------------------------------------------------------------------------
__global__ __launch_bounds__(256)
void gather_kernel(const float* __restrict__ heter,
                   const float* __restrict__ boxes,
                   const float* __restrict__ poi,
                   float* __restrict__ gtOut,
                   float* __restrict__ validArr) {
    const int bn = blockIdx.x;          // 0 .. B*NB-1
    const int b  = bn / NB;
    const int s  = threadIdx.x;

    const float* box = boxes + (size_t)bn * 7;
    const float bx  = box[0];
    const float by  = box[1];
    const float w   = box[4];
    const float l   = box[5];
    const float yaw = box[6];

    float si, co;
    __sincosf(yaw, &si, &co);

    // Tinv translation column
    const float x_oe = -(co * bx + si * by);
    const float y_oe =  si * bx - co * by;

    const int in_left  = (x_oe < -l * 0.5f);
    const int in_right = (x_oe > -l * 0.5f);   // reference uses -l/2 here too
    const int in_up    = (y_oe < -w * 0.5f);
    const int in_down  = (y_oe >  w * 0.5f);

    const size_t point = (size_t)bn * SS + s;
    const float px = poi[point * 2 + 0];
    const float py = poi[point * 2 + 1];

    const int dep = in_left  * (px >  0.6f) + in_right * (px < -0.6f) +
                    in_up    * (py >  0.6f) + in_down  * (py < -0.6f);
    const float vf = (dep <= 1) ? 1.0f : 0.0f;
    validArr[point] = vf;

    // pos_ego = T * [px*l, py*w, 1]
    const float ox = px * l, oy = py * w;
    const float ex = co * ox - si * oy + bx;
    const float ey = si * ox + co * oy + by;
    const float xi = (ex + 102.4f) * 1.25f + 0.5f;   // /0.8
    const float yi = (ey + 102.4f) * 1.25f + 0.5f;

    const int x0i = (int)floorf(xi);
    const int y0i = (int)floorf(yi);
    const int x0 = min(max(x0i,     0), WW - 1);
    const int x1 = min(max(x0i + 1, 0), WW - 1);
    const int y0 = min(max(y0i,     0), HH - 1);
    const int y1 = min(max(y0i + 1, 0), HH - 1);

    const float xf0 = (float)x0, xf1 = (float)x1;
    const float yf0 = (float)y0, yf1 = (float)y1;
    const float wa = (xf1 - xi) * (yf1 - yi);
    const float wb = (xf1 - xi) * (yi - yf0);
    const float wc = (xi - xf0) * (yf1 - yi);
    const float wd = (xi - xf0) * (yi - yf0);

    const float* pbase = heter + (size_t)b * CC * (HH * WW);
    const int iA = y0 * WW + x0;
    const int iB = y1 * WW + x0;
    const int iC = y0 * WW + x1;
    const int iD = y1 * WW + x1;

    float* out = gtOut + point * CC;
    for (int c4 = 0; c4 < CC; c4 += 4) {
        f32x4 r;
        #pragma unroll
        for (int j = 0; j < 4; ++j) {
            const float* pc = pbase + (size_t)(c4 + j) * (HH * WW);
            r[j] = (pc[iA] * wa + pc[iB] * wb + pc[iC] * wc + pc[iD] * wd) * vf;
        }
        *(f32x4*)(out + c4) = r;   // 16B store per lane
    }
}

// ---------------------------------------------------------------------------
// MLP GEMM layer, N = 256, output (bias + ReLU, bf16) to LDS.
// Wave owns 2 column tiles; 8 M-tiles -> 16 f32 16x16 accumulators.
// ---------------------------------------------------------------------------
__device__ __forceinline__
void mlp_layer256(const unsigned short* __restrict__ wt, int Kpad,
                  const float* __restrict__ bias,
                  const unsigned short* __restrict__ inBuf,
                  unsigned short* __restrict__ outBuf,
                  int wave, int lane) {
    v8f acc[8][2];
    #pragma unroll
    for (int mt = 0; mt < 8; ++mt) { acc[mt][0] = zero_v8f(); acc[mt][1] = zero_v8f(); }

    const int ln = lane & 15;
    const int kh = lane >> 4;           // 0: K-low half, 1: K-high half
    const int n0 = wave * 32;

    const unsigned short* wrow0 = wt + (size_t)(n0 + ln) * Kpad + kh * 16;
    const unsigned short* wrow1 = wrow0 + (size_t)16 * Kpad;

    for (int kt = 0; kt < Kpad; kt += 32) {
        FragBf b0, b1;
        b0.f[0] = *(const f32x4*)(wrow0 + kt);
        b0.f[1] = *(const f32x4*)(wrow0 + kt + 8);
        b1.f[0] = *(const f32x4*)(wrow1 + kt);
        b1.f[1] = *(const f32x4*)(wrow1 + kt + 8);
        #pragma unroll
        for (int mt = 0; mt < 8; ++mt) {
            FragBf a;
            const unsigned short* ar = inBuf + (size_t)(mt * 16 + ln) * LDA + kt + kh * 16;
            a.f[0] = *(const f32x4*)(ar);
            a.f[1] = *(const f32x4*)(ar + 8);
            acc[mt][0] = __builtin_amdgcn_wmma_f32_16x16x32_bf16(
                false, a.v, false, b0.v, (short)0, acc[mt][0], false, false);
            acc[mt][1] = __builtin_amdgcn_wmma_f32_16x16x32_bf16(
                false, a.v, false, b1.v, (short)0, acc[mt][1], false, false);
        }
    }

    const float bia0 = bias[n0 + ln];
    const float bia1 = bias[n0 + 16 + ln];
    #pragma unroll
    for (int mt = 0; mt < 8; ++mt) {
        #pragma unroll
        for (int r = 0; r < 8; ++r) {
            const int rowL = mt * 16 + r + (kh << 3);
            float v0 = fmaxf(acc[mt][0][r] + bia0, 0.0f);
            float v1 = fmaxf(acc[mt][1][r] + bia1, 0.0f);
            outBuf[(size_t)rowL * LDA + n0 + ln]      = f2bf(v0);
            outBuf[(size_t)rowL * LDA + n0 + 16 + ln] = f2bf(v1);
        }
    }
}

// Final layer: N = 128, bias (no ReLU), * valid, f32 store to global.
__device__ __forceinline__
void mlp_layer_final(const unsigned short* __restrict__ wt,
                     const float* __restrict__ bias,
                     const unsigned short* __restrict__ inBuf,
                     float* __restrict__ outPred,
                     const float* __restrict__ validArr,
                     int row0, int wave, int lane) {
    v8f acc[8];
    #pragma unroll
    for (int mt = 0; mt < 8; ++mt) acc[mt] = zero_v8f();

    const int ln = lane & 15;
    const int kh = lane >> 4;
    const int n0 = wave * 16;
    const unsigned short* wrow = wt + (size_t)(n0 + ln) * 256 + kh * 16;

    for (int kt = 0; kt < 256; kt += 32) {
        FragBf bf;
        bf.f[0] = *(const f32x4*)(wrow + kt);
        bf.f[1] = *(const f32x4*)(wrow + kt + 8);
        #pragma unroll
        for (int mt = 0; mt < 8; ++mt) {
            FragBf a;
            const unsigned short* ar = inBuf + (size_t)(mt * 16 + ln) * LDA + kt + kh * 16;
            a.f[0] = *(const f32x4*)(ar);
            a.f[1] = *(const f32x4*)(ar + 8);
            acc[mt] = __builtin_amdgcn_wmma_f32_16x16x32_bf16(
                false, a.v, false, bf.v, (short)0, acc[mt], false, false);
        }
    }

    const float bia = bias[n0 + ln];
    #pragma unroll
    for (int mt = 0; mt < 8; ++mt) {
        #pragma unroll
        for (int r = 0; r < 8; ++r) {
            const int grow = row0 + mt * 16 + r + (kh << 3);
            const float val = (acc[mt][r] + bia) * validArr[grow];
            outPred[(size_t)grow * CC + n0 + ln] = val;
        }
    }
}

// ---------------------------------------------------------------------------
// MLP kernel: 128-row tile per block, 8 waves. Sin/cos embedding -> LDS,
// then 6 bf16-WMMA GEMM layers with LDS double buffering.
// ---------------------------------------------------------------------------
__global__ __launch_bounds__(256)
void mlp_kernel(const float* __restrict__ poi,
                const float* __restrict__ validArr,
                const unsigned short* __restrict__ wts,
                const float* __restrict__ b0, const float* __restrict__ b1,
                const float* __restrict__ b2, const float* __restrict__ b3,
                const float* __restrict__ b4, const float* __restrict__ b5,
                float* __restrict__ outPred) {
    __shared__ unsigned short bufA[128 * LDA];
    __shared__ unsigned short bufB[128 * LDA];

    const int row0 = blockIdx.x * 128;
    const int tid  = threadIdx.x;
    const int wave = tid >> 5;
    const int lane = tid & 31;

    // ---- embedding: 34 features, zero-padded to K=64, bf16 into bufA ----
    if (tid < 128) {
        const int row = row0 + tid;
        const float px = poi[(size_t)row * 2 + 0];
        const float py = poi[(size_t)row * 2 + 1];
        unsigned short* er = bufA + (size_t)tid * LDA;
        er[0] = f2bf(px);
        er[1] = f2bf(py);
        #pragma unroll
        for (int j = 0; j < 8; ++j) {
            const float f = (float)(1 << j);
            float s0, c0, s1, c1;
            __sincosf(f * px, &s0, &c0);
            __sincosf(f * py, &s1, &c1);
            er[2 + 4 * j + 0] = f2bf(s0);
            er[2 + 4 * j + 1] = f2bf(s1);
            er[2 + 4 * j + 2] = f2bf(c0);
            er[2 + 4 * j + 3] = f2bf(c1);
        }
        #pragma unroll
        for (int k = 34; k < 64; ++k) er[k] = 0;
    }
    __syncthreads();

    mlp_layer256(wts + OFF_W0,  64, b0, bufA, bufB, wave, lane); __syncthreads();
    mlp_layer256(wts + OFF_W1, 256, b1, bufB, bufA, wave, lane); __syncthreads();
    mlp_layer256(wts + OFF_W2, 256, b2, bufA, bufB, wave, lane); __syncthreads();
    mlp_layer256(wts + OFF_W3, 256, b3, bufB, bufA, wave, lane); __syncthreads();
    mlp_layer256(wts + OFF_W4, 256, b4, bufA, bufB, wave, lane); __syncthreads();
    mlp_layer_final(wts + OFF_W5, b5, bufB, outPred, validArr, row0, wave, lane);
}

// ---------------------------------------------------------------------------
// Host launch
// ---------------------------------------------------------------------------
extern "C" void kernel_launch(void* const* d_in, const int* in_sizes, int n_in,
                              void* d_out, int out_size, void* d_ws, size_t ws_size,
                              hipStream_t stream) {
    (void)in_sizes; (void)n_in; (void)out_size; (void)ws_size;

    const float* heter = (const float*)d_in[0];
    const float* boxes = (const float*)d_in[1];
    const float* poi   = (const float*)d_in[2];
    const float* w0 = (const float*)d_in[3];
    const float* b0 = (const float*)d_in[4];
    const float* w1 = (const float*)d_in[5];
    const float* b1 = (const float*)d_in[6];
    const float* w2 = (const float*)d_in[7];
    const float* b2 = (const float*)d_in[8];
    const float* w3 = (const float*)d_in[9];
    const float* b3 = (const float*)d_in[10];
    const float* w4 = (const float*)d_in[11];
    const float* b4 = (const float*)d_in[12];
    const float* w5 = (const float*)d_in[13];
    const float* b5 = (const float*)d_in[14];

    float* out      = (float*)d_out;
    float* outPred  = out + (size_t)HETER_ELEMS;
    float* outGt    = outPred + (size_t)PRED_ELEMS;

    // Scratch: valid mask (f32, MTOT), then bf16 transposed weights
    float* validArr = (float*)d_ws;
    unsigned short* wts = (unsigned short*)((char*)d_ws + (size_t)MTOT * sizeof(float));

    // 1) heter passthrough — pure bandwidth, use the copy engine path
    hipMemcpyAsync(d_out, d_in[0], (size_t)HETER_ELEMS * sizeof(float),
                   hipMemcpyDeviceToDevice, stream);

    // 2) weight prep (tiny): f32 [K][N] -> bf16 [N][Kpad]
    {
        struct { const float* src; int K, N, Kpad, off; } jobs[6] = {
            { w0,  34, 256,  64, OFF_W0 },
            { w1, 256, 256, 256, OFF_W1 },
            { w2, 256, 256, 256, OFF_W2 },
            { w3, 256, 256, 256, OFF_W3 },
            { w4, 256, 256, 256, OFF_W4 },
            { w5, 256, 128, 256, OFF_W5 },
        };
        for (int i = 0; i < 6; ++i) {
            const int cnt = jobs[i].N * jobs[i].Kpad;
            convert_weight_kernel<<<(cnt + 255) / 256, 256, 0, stream>>>(
                jobs[i].src, wts + jobs[i].off, jobs[i].K, jobs[i].N, jobs[i].Kpad);
        }
    }

    // 3) bilinear gather (gt) + valid mask
    gather_kernel<<<BB * NB, 256, 0, stream>>>(heter, boxes, poi, outGt, validArr);

    // 4) MLP (pred) via bf16 WMMA
    mlp_kernel<<<MTOT / 128, 256, 0, stream>>>(poi, validArr, wts,
                                               b0, b1, b2, b3, b4, b5, outPred);
}
